// MHCLayer_85005992722500
// MI455X (gfx1250) — matrix operand order repaired and checked by reading, not compiled, not run on hardware
//
#include <hip/hip_runtime.h>
#include <hip/hip_bf16.h>
#include <math.h>

typedef __attribute__((ext_vector_type(16))) __bf16 v16bf;
typedef __attribute__((ext_vector_type(8)))  float  v8f;

#define TOK  8192     // B*L
#define CDIM 1024     // C
#define NS   4        // streams
#define HID  4096     // 4*C
#define NCOL 24       // n*n + 2n

// LDS byte-offset of a __shared__ object (VDST operand of async-to-LDS ops).
__device__ __forceinline__ unsigned lds_off(const void* p) {
    return (unsigned)(unsigned long long)
        (__attribute__((address_space(3))) const void*)p;
}

// global_load_async_to_lds_b128: per-lane 16B, LDS dest from VGPR, ASYNCcnt.
// INST_OFFSET is added to BOTH the LDS and the global address.
#define GLD_ASYNC_B128(ldsoff, gaddr, OFF)                                    \
    asm volatile("global_load_async_to_lds_b128 %0, %1, off offset:" OFF     \
                 :: "v"(ldsoff), "v"(gaddr) : "memory")

#define WAIT_ASYNC0() asm volatile("s_wait_asynccnt 0x0" ::: "memory")

// ---------------------------------------------------------------------------
// Kernel 1: per-token dynamics. One block (256 threads) per token.
// ---------------------------------------------------------------------------
__global__ __launch_bounds__(256) void k_token_dyn(
    const float* __restrict__ x,
    const float* __restrict__ rmsw,
    const float* __restrict__ Wdyn,
    const float* __restrict__ bias_pre,
    const float* __restrict__ bias_post,
    const float* __restrict__ bias_res,
    const float* __restrict__ a_pre,
    const float* __restrict__ a_post,
    const float* __restrict__ a_res,
    __bf16* __restrict__ pre_mixed,
    float*  __restrict__ params)
{
    __shared__ float xl[CDIM];
    __shared__ float red[256];
    __shared__ float red24[256 * NCOL];
    __shared__ float spre_s;

    const int t   = blockIdx.x;
    const int tid = threadIdx.x;
    const float* xrow = x + (size_t)t * CDIM;

    float ss = 0.f;
    for (int c = tid; c < CDIM; c += 256) { float v = xrow[c]; xl[c] = v; ss += v * v; }
    red[tid] = ss;
    __syncthreads();
    for (int s = 128; s > 0; s >>= 1) {
        if (tid < s) red[tid] += red[tid + s];
        __syncthreads();
    }
    // mean over 4x-repeated vector == mean over C
    const float inv_rms = 1.0f / sqrtf(red[0] * (1.0f / (float)CDIM) + 1e-8f);

    float acc[NCOL];
#pragma unroll
    for (int j = 0; j < NCOL; j++) acc[j] = 0.f;
    for (int it = 0; it < (NS * CDIM) / 256; it++) {
        const int k  = it * 256 + tid;
        const float xn = xl[k & (CDIM - 1)] * inv_rms * rmsw[k];
        const float* wr = Wdyn + (size_t)k * NCOL;
#pragma unroll
        for (int j = 0; j < NCOL; j++) acc[j] = fmaf(xn, wr[j], acc[j]);
    }
#pragma unroll
    for (int j = 0; j < NCOL; j++) red24[tid * NCOL + j] = acc[j];
    __syncthreads();
    for (int s = 128; s > 0; s >>= 1) {
        if (tid < s) {
#pragma unroll
            for (int j = 0; j < NCOL; j++)
                red24[tid * NCOL + j] += red24[(tid + s) * NCOL + j];
        }
        __syncthreads();
    }

    if (tid == 0) {
        float dyn[NCOL];
#pragma unroll
        for (int j = 0; j < NCOL; j++) dyn[j] = red24[j];
        const float ap = a_pre[0], apo = a_post[0], ar = a_res[0];
        float spre = 0.f, hpost[NS];
#pragma unroll
        for (int i = 0; i < NS; i++) {
            spre     += 1.0f / (1.0f + expf(-(ap  * dyn[i]      + bias_pre[i])));
            hpost[i]  = 2.0f / (1.0f + expf(-(apo * dyn[NS + i] + bias_post[i])));
        }
        float m[NS][NS];
#pragma unroll
        for (int i = 0; i < NS; i++)
#pragma unroll
            for (int j = 0; j < NS; j++)
                m[i][j] = expf(ar * dyn[2 * NS + i * NS + j] + bias_res[i * NS + j]);
        for (int it = 0; it < 20; it++) {
#pragma unroll
            for (int j = 0; j < NS; j++) {
                float cs = m[0][j] + m[1][j] + m[2][j] + m[3][j] + 1e-8f;
                float ics = 1.0f / cs;
#pragma unroll
                for (int i = 0; i < NS; i++) m[i][j] *= ics;
            }
#pragma unroll
            for (int i = 0; i < NS; i++) {
                float rs = m[i][0] + m[i][1] + m[i][2] + m[i][3] + 1e-8f;
                float irs = 1.0f / rs;
#pragma unroll
                for (int j = 0; j < NS; j++) m[i][j] *= irs;
            }
        }
#pragma unroll
        for (int i = 0; i < NS; i++) {
            params[(size_t)t * 8 + i]      = hpost[i];
            params[(size_t)t * 8 + 4 + i]  = m[i][0] + m[i][1] + m[i][2] + m[i][3];
        }
        spre_s = spre;
    }
    __syncthreads();
    const float sp = spre_s;
    for (int c = tid; c < CDIM; c += 256)
        pre_mixed[(size_t)t * CDIM + c] = (__bf16)(sp * xl[c]);
}

// ---------------------------------------------------------------------------
// Kernel 2: weight convert f32 -> bf16 with transpose: Wt[n*K+k] = W[k*N+n].
// ---------------------------------------------------------------------------
__global__ __launch_bounds__(256) void k_convert_w(
    const float* __restrict__ W, __bf16* __restrict__ Wt, int K, int N)
{
    int idx = blockIdx.x * 256 + threadIdx.x;
    if (idx < K * N) {
        int k = idx / N, n = idx - k * N;
        Wt[(size_t)n * K + k] = (__bf16)W[idx];
    }
}

// ---------------------------------------------------------------------------
// Kernel 3: bf16 WMMA GEMM, async-to-LDS double-buffered, BK = 64.
// C[M,N] = A[M,K] * Bt[N,K]^T (+ epilogue).
// 128x128 block tile, 256 threads = 8 waves, wave computes 32x64 (2x4 tiles).
// MODE 0: bf16 out = gelu(acc + bias); MODE 1: f32 acc+bias; MODE 2: +xres.
// ---------------------------------------------------------------------------
template <int MODE>
__global__ __launch_bounds__(256) void k_gemm(
    const __bf16* __restrict__ A,
    const __bf16* __restrict__ Bt,
    const float*  __restrict__ bias,
    void*         __restrict__ out,
    const float*  __restrict__ xres,
    int M, int N, int K)
{
    __shared__ alignas(16) __bf16 As[2][128 * 64];
    __shared__ alignas(16) __bf16 Bs[2][128 * 64];

    const int tid  = threadIdx.x;
    const int m0   = blockIdx.y * 128;
    const int n0   = blockIdx.x * 128;
    const int lane = tid & 31;
    const int wave = tid >> 5;
    const int wr   = wave & 3;   // row group: 32 rows
    const int wc   = wave >> 2;  // col group: 64 cols
    const int l16  = lane & 15;
    const int lhi  = lane >> 4;  // 0/1

    v8f accv[2][4];
#pragma unroll
    for (int rt = 0; rt < 2; rt++)
#pragma unroll
        for (int ct = 0; ct < 4; ct++) accv[rt][ct] = (v8f)0.0f;

    // Staging: thread t moves row t>>1, k-chunk (t&1)*32 .. +31 (64 bytes).
    const int srow = tid >> 1;
    const int scol = (tid & 1) * 32;
    const unsigned asOff0 = lds_off(&As[0][srow * 64 + scol]);
    const unsigned asOff1 = lds_off(&As[1][srow * 64 + scol]);
    const unsigned bsOff0 = lds_off(&Bs[0][srow * 64 + scol]);
    const unsigned bsOff1 = lds_off(&Bs[1][srow * 64 + scol]);
    const __bf16* gaBase = A  + (size_t)(m0 + srow) * K + scol;
    const __bf16* gbBase = Bt + (size_t)(n0 + srow) * K + scol;

    auto stage = [&](int buf, int k0) {
        const unsigned ao = buf ? asOff1 : asOff0;
        const unsigned bo = buf ? bsOff1 : bsOff0;
        unsigned long long ga = (unsigned long long)(gaBase + k0);
        unsigned long long gb = (unsigned long long)(gbBase + k0);
        GLD_ASYNC_B128(ao, ga, "0");
        GLD_ASYNC_B128(ao, ga, "16");
        GLD_ASYNC_B128(ao, ga, "32");
        GLD_ASYNC_B128(ao, ga, "48");
        GLD_ASYNC_B128(bo, gb, "0");
        GLD_ASYNC_B128(bo, gb, "16");
        GLD_ASYNC_B128(bo, gb, "32");
        GLD_ASYNC_B128(bo, gb, "48");
    };

    union Frag { v16bf v; uint4 q[2]; };

    stage(0, 0);
    WAIT_ASYNC0();
    __syncthreads();

    for (int k0 = 0; k0 < K; k0 += 64) {
        const int buf = (k0 >> 6) & 1;
        if (k0 + 64 < K) stage(buf ^ 1, k0 + 64);  // overlap with compute

#pragma unroll
        for (int ks = 0; ks < 2; ks++) {
            // A fragment (16x32): lanes 0-15 -> K {0..7,16..23}, 16-31 -> {8..15,24..31}
            Frag a[2];
#pragma unroll
            for (int rt = 0; rt < 2; rt++) {
                const __bf16* p = &As[buf][(wr * 32 + rt * 16 + l16) * 64 + ks * 32 + lhi * 8];
                a[rt].q[0] = *(const uint4*)p;
                a[rt].q[1] = *(const uint4*)(p + 16);
            }
            // B fragment (32x16): lane -> N, lanes 0-15 -> K {0..7,16..23}, 16-31 -> {8..15,24..31}
            Frag b[4];
#pragma unroll
            for (int ct = 0; ct < 4; ct++) {
                const __bf16* p = &Bs[buf][(wc * 64 + ct * 16 + l16) * 64 + ks * 32 + lhi * 8];
                b[ct].q[0] = *(const uint4*)p;
                b[ct].q[1] = *(const uint4*)(p + 16);
            }
#pragma unroll
            for (int rt = 0; rt < 2; rt++)
#pragma unroll
                for (int ct = 0; ct < 4; ct++)
                    accv[rt][ct] = __builtin_amdgcn_wmma_f32_16x16x32_bf16(
                        false, a[rt].v, false, b[ct].v,
                        (short)0, accv[rt][ct], false, false);
        }

        WAIT_ASYNC0();       // own async copies for buf^1 have landed
        __syncthreads();     // everyone's landed; everyone done reading buf
    }

    // Epilogue. C/D layout: VGPR i, lanes 0-15 -> M=i, lanes 16-31 -> M=i+8; N=l16.
#pragma unroll
    for (int rt = 0; rt < 2; rt++) {
#pragma unroll
        for (int ct = 0; ct < 4; ct++) {
            const int gn = n0 + wc * 64 + ct * 16 + l16;
            const float bv = bias[gn];
#pragma unroll
            for (int i = 0; i < 8; i++) {
                const int gr = m0 + wr * 32 + rt * 16 + i + lhi * 8;
                const size_t off = (size_t)gr * N + gn;
                float v = accv[rt][ct][i] + bv;
                if (MODE == 0) {
                    v = 0.5f * v * (1.0f + erff(v * 0.70710678118654752f));
                    ((__bf16*)out)[off] = (__bf16)v;
                } else if (MODE == 1) {
                    ((float*)out)[off] = v;
                } else {
                    ((float*)out)[off] = v + xres[off];
                }
            }
        }
    }
}

// ---------------------------------------------------------------------------
// Kernel 4: merged[t, n*C+c] = bf16( r[n]*x[t,c] + H_post[n]*f_out[t,c] )
// ---------------------------------------------------------------------------
__global__ __launch_bounds__(256) void k_merged(
    const float* __restrict__ x,
    const float* __restrict__ f_out,
    const float* __restrict__ params,
    __bf16* __restrict__ merged)
{
    const int t = blockIdx.x, tid = threadIdx.x;
    __shared__ float hp[NS], rr[NS];
    if (tid < 8) {
        float v = params[(size_t)t * 8 + tid];
        if (tid < 4) hp[tid] = v; else rr[tid - 4] = v;
    }
    __syncthreads();
    for (int c = tid; c < CDIM; c += 256) {
        const float xv = x[(size_t)t * CDIM + c];
        const float fv = f_out[(size_t)t * CDIM + c];
#pragma unroll
        for (int n = 0; n < NS; n++)
            merged[(size_t)t * (NS * CDIM) + (size_t)n * CDIM + c] =
                (__bf16)(rr[n] * xv + hp[n] * fv);
    }
}

// ---------------------------------------------------------------------------
extern "C" void kernel_launch(void* const* d_in, const int* in_sizes, int n_in,
                              void* d_out, int out_size, void* d_ws, size_t ws_size,
                              hipStream_t stream) {
    const float* x         = (const float*)d_in[0];
    const float* rms_w     = (const float*)d_in[1];
    const float* W_dyn     = (const float*)d_in[2];
    const float* bias_pre  = (const float*)d_in[3];
    const float* bias_post = (const float*)d_in[4];
    const float* bias_res  = (const float*)d_in[5];
    const float* alpha_pre = (const float*)d_in[6];
    const float* alpha_post= (const float*)d_in[7];
    const float* alpha_res = (const float*)d_in[8];
    const float* W1        = (const float*)d_in[9];
    const float* b1        = (const float*)d_in[10];
    const float* W2        = (const float*)d_in[11];
    const float* b2        = (const float*)d_in[12];
    const float* Wout      = (const float*)d_in[13];
    const float* bout      = (const float*)d_in[14];
    float* out = (float*)d_out;

    char* ws = (char*)d_ws;
    const size_t MB = 1024ull * 1024ull;
    __bf16* W1t    = (__bf16*)(ws + 0 * MB);    //  8 MB  (HID, C)
    __bf16* W2t    = (__bf16*)(ws + 8 * MB);    //  8 MB  (C, HID)
    __bf16* Woutt  = (__bf16*)(ws + 16 * MB);   //  8 MB  (C, NS*C)
    __bf16* premix = (__bf16*)(ws + 24 * MB);   // 16 MB  (TOK, C)
    __bf16* h      = (__bf16*)(ws + 40 * MB);   // 64 MB  (TOK, HID)
    float*  f_out  = (float*) (ws + 104 * MB);  // 32 MB  (TOK, C)
    __bf16* merged = (__bf16*)(ws + 136 * MB);  // 64 MB  (TOK, NS*C)
    float*  params = (float*) (ws + 200 * MB);  // 256 KB (TOK, 8)

    // 1) weight convert + transpose (f32 -> bf16, (K,N) -> (N,K))
    {
        int n1 = CDIM * HID;         // W1 (1024,4096)
        k_convert_w<<<(n1 + 255) / 256, 256, 0, stream>>>(W1, W1t, CDIM, HID);
        int n2 = HID * CDIM;         // W2 (4096,1024)
        k_convert_w<<<(n2 + 255) / 256, 256, 0, stream>>>(W2, W2t, HID, CDIM);
        int n3 = (NS * CDIM) * CDIM; // Wout (4096,1024)
        k_convert_w<<<(n3 + 255) / 256, 256, 0, stream>>>(Wout, Woutt, NS * CDIM, CDIM);
    }

    // 2) per-token gates + pre_mixed
    k_token_dyn<<<TOK, 256, 0, stream>>>(x, rms_w, W_dyn, bias_pre, bias_post,
                                         bias_res, alpha_pre, alpha_post,
                                         alpha_res, premix, params);

    // 3) h = gelu(pre_mixed @ W1 + b1)   [M=8192, N=4096, K=1024]
    k_gemm<0><<<dim3(HID / 128, TOK / 128), 256, 0, stream>>>(
        premix, W1t, b1, (void*)h, nullptr, TOK, HID, CDIM);

    // 4) f_out = h @ W2 + b2             [M=8192, N=1024, K=4096]
    k_gemm<1><<<dim3(CDIM / 128, TOK / 128), 256, 0, stream>>>(
        h, W2t, b2, (void*)f_out, nullptr, TOK, CDIM, HID);

    // 5) merged = r[n]*x + H_post[n]*f_out
    k_merged<<<TOK, 256, 0, stream>>>(x, f_out, params, merged);

    // 6) out = merged @ Wout + bout + x  [M=8192, N=1024, K=4096]
    k_gemm<2><<<dim3(CDIM / 128, TOK / 128), 256, 0, stream>>>(
        merged, Woutt, bout, (void*)out, x, TOK, CDIM, NS * CDIM);
}
